// mainNetwork_16741782519916
// MI455X (gfx1250) — matrix operand order repaired and checked by reading, not compiled, not run on hardware
//
#include <hip/hip_runtime.h>

// ---------------------------------------------------------------------------
// Types for CDNA5 WMMA (wave32, v_wmma_f32_16x16x32_bf16)
// ---------------------------------------------------------------------------
typedef __attribute__((ext_vector_type(16))) __bf16 v16bf;
typedef __attribute__((ext_vector_type(8)))  __bf16 v8bf;
typedef __attribute__((ext_vector_type(4)))  __bf16 v4bf;
typedef __attribute__((ext_vector_type(8)))  float  v8f;
typedef __attribute__((ext_vector_type(4)))  float  v4f;

// ---------------------------------------------------------------------------
// Implicit-GEMM parameters. Covers:
//   - conv3d (stride 1, arbitrary kernel/padding), weights row-major [Co][Ci*kd*kh*kw]
//   - plain GEMM (set KD=KH=KW=1, Din=Hin=1, Win=N)
//   - convT(1,2,2)   (Alayout=1: A[m][k] = W[(k*M+m)*4 + ij], strided output)
// ---------------------------------------------------------------------------
struct GemmP {
  const float* A; const float* B; const float* bias; float* C;
  int M, N, K;
  int Alayout, ij;               // 0: row-major; 1: convT weight (Cin,Cout,2,2)
  int KD, KH, KW, pd, ph, pw;    // im2col kernel + padding
  int Din, Hin, Win;             // input spatial dims
  int Dn, Hn, Wn;                // GEMM-N decomposition (output spatial dims)
  int oH, oW, sy, sx, offy, offx;// output mapping (convT stride/offset support)
};

// Block tile: 128 (M) x 64 (N). 8 waves, one 16-row M-slice per wave,
// 4 N-subtiles (4 accumulators) per wave -> 4 WMMAs per wave per K-chunk
// reusing a single A fragment.
__global__ __launch_bounds__(256) void wmma_gemm_conv(GemmP p) {
  __shared__ __align__(32) __bf16 As[128 * 32];   // [m][k]  8 KB
  __shared__ __align__(32) __bf16 Bs[64 * 32];    // [n][k]  4 KB

  const int tid  = threadIdx.x;
  const int wave = tid >> 5;
  const int lane = tid & 31;
  const int hf   = lane >> 4;     // 0: lanes 0-15, 1: lanes 16-31
  const int lo   = lane & 15;

  const int m0 = blockIdx.y * 128;
  const int n0 = blockIdx.x * 64;

  const int KHW  = p.KH * p.KW;
  const int KVOL = p.KD * KHW;
  const int HWn  = p.Hn * p.Wn;

  // Fast staging predicates (uniform across the block)
  const bool fastA = (p.Alayout == 0) && ((p.K & 3) == 0);
  const bool fastB = (KVOL == 1) && ((p.pd | p.ph | p.pw) == 0) && ((p.N & 3) == 0);

  v8f acc0 = {0.f,0.f,0.f,0.f,0.f,0.f,0.f,0.f};
  v8f acc1 = acc0, acc2 = acc0, acc3 = acc0;

  for (int k0 = 0; k0 < p.K; k0 += 32) {
    __syncthreads();
    // ---- stage A tile: 128 rows x 32 k (bf16) ----
    if (fastA) {
      // 4 contiguous k per thread: b128 global load -> packed bf16 -> b64 LDS store
      for (int t = tid; t < 1024; t += 256) {
        int ml = t >> 3, kq = (t & 7) * 4;
        int gm = m0 + ml, gk = k0 + kq;
        v4bf d = {(__bf16)0.f, (__bf16)0.f, (__bf16)0.f, (__bf16)0.f};
        if (gm < p.M && gk < p.K) {
          const float* src = &p.A[(long long)gm * p.K + gk];
          v4f f = *(const v4f*)src;
          d[0] = (__bf16)f[0]; d[1] = (__bf16)f[1];
          d[2] = (__bf16)f[2]; d[3] = (__bf16)f[3];
          __builtin_prefetch(src + 32, 0, 1);  // next K-chunk (global_prefetch_b8)
        }
        *(v4bf*)&As[ml * 32 + kq] = d;
      }
    } else if (p.Alayout == 0) {
      for (int t = tid; t < 128 * 32; t += 256) {
        int ml = t >> 5, kl = t & 31;
        int gm = m0 + ml, gk = k0 + kl;
        float v = (gm < p.M && gk < p.K) ? p.A[(long long)gm * p.K + gk] : 0.f;
        As[t] = (__bf16)v;
      }
    } else {
      for (int t = tid; t < 128 * 32; t += 256) {
        int ml = t >> 5, kl = t & 31;
        int gm = m0 + ml, gk = k0 + kl;
        float v = (gm < p.M && gk < p.K)
                      ? p.A[((long long)gk * p.M + gm) * 4 + p.ij] : 0.f;
        As[t] = (__bf16)v;
      }
    }
    // ---- stage B tile: 64 n x 32 k (bf16) ----
    if (fastB) {
      // address reduces to B[k*N + n]; n is memory-contiguous -> b128 loads
      for (int t = tid; t < 512; t += 256) {
        int kl = t >> 4, ng = (t & 15) * 4;
        int gk = k0 + kl, gn = n0 + ng;
        float f0 = 0.f, f1 = 0.f, f2 = 0.f, f3 = 0.f;
        if (gk < p.K && gn < p.N) {   // N % 4 == 0 -> whole group in-bounds
          v4f f = *(const v4f*)&p.B[(long long)gk * p.N + gn];
          f0 = f[0]; f1 = f[1]; f2 = f[2]; f3 = f[3];
        }
        Bs[(ng + 0) * 32 + kl] = (__bf16)f0;
        Bs[(ng + 1) * 32 + kl] = (__bf16)f1;
        Bs[(ng + 2) * 32 + kl] = (__bf16)f2;
        Bs[(ng + 3) * 32 + kl] = (__bf16)f3;
      }
    } else {
      for (int t = tid; t < 64 * 32; t += 256) {
        int nl = t >> 5, kl = t & 31;
        int gn = n0 + nl, gk = k0 + kl;
        float v = 0.f;
        if (gn < p.N && gk < p.K) {
          int ci = gk / KVOL; int r = gk - ci * KVOL;
          int kd = r / KHW;   r -= kd * KHW;
          int kh = r / p.KW;  int kw = r - kh * p.KW;
          int z  = gn / HWn;  int r2 = gn - z * HWn;
          int y  = r2 / p.Wn; int x  = r2 - y * p.Wn;
          int iz = z + kd - p.pd, iy = y + kh - p.ph, ix = x + kw - p.pw;
          if (iz >= 0 && iz < p.Din && iy >= 0 && iy < p.Hin && ix >= 0 && ix < p.Win)
            v = p.B[(((long long)ci * p.Din + iz) * p.Hin + iy) * p.Win + ix];
        }
        Bs[t] = (__bf16)v;
      }
    }
    __syncthreads();

    // ---- A fragment, contiguous k order (same as B) ----
    // lane hf=0 -> K 0..15; hf=1 -> K 16..31
    int mrow = wave * 16 + lo;
    const v8bf* ap = (const v8bf*)&As[mrow * 32 + hf * 16];
    v8bf a_lo = ap[0];
    v8bf a_hi = ap[1];
    v16bf a;
#pragma unroll
    for (int i = 0; i < 8; ++i) { a[i] = a_lo[i]; a[i + 8] = a_hi[i]; }

    // ---- 4 B fragments / 4 WMMAs reusing the same A ----
    // B (32x16): lane hf=0 -> K 0..15; hf=1 -> K 16..31, n = lo
    v16bf b0 = *(const v16bf*)&Bs[(0 * 16 + lo) * 32 + hf * 16];
    v16bf b1 = *(const v16bf*)&Bs[(1 * 16 + lo) * 32 + hf * 16];
    v16bf b2 = *(const v16bf*)&Bs[(2 * 16 + lo) * 32 + hf * 16];
    v16bf b3 = *(const v16bf*)&Bs[(3 * 16 + lo) * 32 + hf * 16];
    acc0 = __builtin_amdgcn_wmma_f32_16x16x32_bf16(false, a, false, b0, (short)0, acc0, false, false);
    acc1 = __builtin_amdgcn_wmma_f32_16x16x32_bf16(false, a, false, b1, (short)0, acc1, false, false);
    acc2 = __builtin_amdgcn_wmma_f32_16x16x32_bf16(false, a, false, b2, (short)0, acc2, false, false);
    acc3 = __builtin_amdgcn_wmma_f32_16x16x32_bf16(false, a, false, b3, (short)0, acc3, false, false);
  }

  // ---- store 4x 16x16 f32 tiles (C/D layout: VGPR v -> M = v + 8*hf, N = lo)
  int gmBase = m0 + wave * 16 + hf * 8;
  float bias_cache[8];
#pragma unroll
  for (int v = 0; v < 8; ++v) {
    int gm = gmBase + v;
    bias_cache[v] = (p.bias && gm < p.M) ? p.bias[gm] : 0.f;
  }
#pragma unroll
  for (int nsub = 0; nsub < 4; ++nsub) {
    v8f acc = (nsub == 0) ? acc0 : (nsub == 1) ? acc1 : (nsub == 2) ? acc2 : acc3;
    int gn = n0 + nsub * 16 + lo;
    if (gn < p.N) {
      int z  = gn / HWn;  int r2 = gn - z * HWn;
      int y  = r2 / p.Wn; int x  = r2 - y * p.Wn;
      int yy = y * p.sy + p.offy;
      int xx = x * p.sx + p.offx;
#pragma unroll
      for (int v = 0; v < 8; ++v) {
        int gm = gmBase + v;
        if (gm < p.M)
          p.C[(((long long)gm * p.Dn + z) * p.oH + yy) * p.oW + xx] = acc[v] + bias_cache[v];
      }
    }
  }
}

// ---------------------------------------------------------------------------
// GroupNorm: per-group reduce (one block per group), then apply (+LeakyReLU)
// ---------------------------------------------------------------------------
__global__ __launch_bounds__(256) void gn_reduce(const float* x, float* stats,
                                                 int C, long long S, int groups) {
  int g = blockIdx.x;
  int cpg = C / groups;
  long long cnt = (long long)cpg * S;
  const float* p = x + (long long)g * cnt;
  float s = 0.f, q = 0.f;
  for (long long i = threadIdx.x; i < cnt; i += 256) { float v = p[i]; s += v; q += v * v; }
  __shared__ float sh[512];
  sh[threadIdx.x] = s; sh[256 + threadIdx.x] = q;
  __syncthreads();
  for (int o = 128; o > 0; o >>= 1) {
    if (threadIdx.x < o) {
      sh[threadIdx.x] += sh[threadIdx.x + o];
      sh[256 + threadIdx.x] += sh[256 + threadIdx.x + o];
    }
    __syncthreads();
  }
  if (threadIdx.x == 0) {
    float mean = sh[0] / (float)cnt;
    float var  = sh[256] / (float)cnt - mean * mean;
    stats[g * 2] = mean; stats[g * 2 + 1] = var;
  }
}

__global__ __launch_bounds__(256) void gn_apply(const float* x, float* y,
                                                const float* stats, const float* gm,
                                                const float* bt, int C, long long S,
                                                int groups, int lrelu) {
  long long i = (long long)blockIdx.x * 256 + threadIdx.x;
  long long total = (long long)C * S;
  if (i >= total) return;
  int c = (int)(i / S);
  int g = c / (C / groups);
  float mean = stats[2 * g], var = stats[2 * g + 1];
  float v = (x[i] - mean) * rsqrtf(var + 1e-5f) * gm[c] + bt[c];
  if (lrelu && v < 0.f) v *= 0.01f;
  y[i] = v;
}

// ---------------------------------------------------------------------------
// MaxPool3d (1,2,2)
// ---------------------------------------------------------------------------
__global__ __launch_bounds__(256) void maxpool_1x2x2(const float* x, float* y,
                                                     long long total, int D, int H, int W) {
  long long i = (long long)blockIdx.x * 256 + threadIdx.x;
  if (i >= total) return;
  int Ho = H >> 1, Wo = W >> 1;
  long long wo = i % Wo; long long r = i / Wo;
  long long ho = r % Ho; r /= Ho;
  long long d  = r % D;  long long c = r / D;
  const float* p = x + ((c * D + d) * H + ho * 2) * W + wo * 2;
  y[i] = fmaxf(fmaxf(p[0], p[1]), fmaxf(p[W], p[W + 1]));
}

__global__ __launch_bounds__(256) void add2_kernel(float* dst, const float* a,
                                                   const float* b, long long n) {
  long long i = (long long)blockIdx.x * 256 + threadIdx.x;
  if (i < n) dst[i] = a[i] + b[i];
}

// ---------------------------------------------------------------------------
// Row softmax (in place), one block per row
// ---------------------------------------------------------------------------
__global__ __launch_bounds__(256) void softmax_rows(float* m, int cols) {
  float* p = m + (long long)blockIdx.x * cols;
  __shared__ float sh[256];
  float mx = -3.4e38f;
  for (int i = threadIdx.x; i < cols; i += 256) mx = fmaxf(mx, p[i]);
  sh[threadIdx.x] = mx; __syncthreads();
  for (int o = 128; o > 0; o >>= 1) {
    if (threadIdx.x < o) sh[threadIdx.x] = fmaxf(sh[threadIdx.x], sh[threadIdx.x + o]);
    __syncthreads();
  }
  mx = sh[0]; __syncthreads();
  float sum = 0.f;
  for (int i = threadIdx.x; i < cols; i += 256) { float e = __expf(p[i] - mx); p[i] = e; sum += e; }
  sh[threadIdx.x] = sum; __syncthreads();
  for (int o = 128; o > 0; o >>= 1) {
    if (threadIdx.x < o) sh[threadIdx.x] += sh[threadIdx.x + o];
    __syncthreads();
  }
  float inv = 1.f / sh[0];
  for (int i = threadIdx.x; i < cols; i += 256) p[i] *= inv;
}

// Final channel softmax over 4 channels
__global__ __launch_bounds__(256) void softmax_c4(const float* x, float* y, long long S) {
  long long i = (long long)blockIdx.x * 256 + threadIdx.x;
  if (i >= S) return;
  float a0 = x[i], a1 = x[S + i], a2 = x[2 * S + i], a3 = x[3 * S + i];
  float mx = fmaxf(fmaxf(a0, a1), fmaxf(a2, a3));
  float e0 = __expf(a0 - mx), e1 = __expf(a1 - mx), e2 = __expf(a2 - mx), e3 = __expf(a3 - mx);
  float inv = 1.f / (e0 + e1 + e2 + e3);
  y[i] = e0 * inv; y[S + i] = e1 * inv; y[2 * S + i] = e2 * inv; y[3 * S + i] = e3 * inv;
}

// ---------------------------------------------------------------------------
// Host-side helpers
// ---------------------------------------------------------------------------
namespace {

struct WS { char* base; size_t cur, cap; };
static float* wsalloc(WS& w, long long nfloats) {
  float* p = (float*)(w.base + w.cur);
  w.cur += (((size_t)nfloats * 4) + 255) & ~(size_t)255;
  return p;
}

static void launch_gemm(hipStream_t s, GemmP& p) {
  dim3 grid((p.N + 63) / 64, (p.M + 127) / 128);
  wmma_gemm_conv<<<grid, 256, 0, s>>>(p);
}

static void conv3d(hipStream_t s, const float* x, int Ci, int D, int H, int W,
                   const float* w, const float* b, int Co,
                   int kd, int kh, int kw, int pd, int ph, int pw, float* y) {
  GemmP p{};
  p.A = w; p.B = x; p.bias = b; p.C = y;
  p.M = Co; p.N = D * H * W; p.K = Ci * kd * kh * kw;
  p.Alayout = 0; p.ij = 0;
  p.KD = kd; p.KH = kh; p.KW = kw; p.pd = pd; p.ph = ph; p.pw = pw;
  p.Din = D; p.Hin = H; p.Win = W;
  p.Dn = D; p.Hn = H; p.Wn = W;
  p.oH = H; p.oW = W; p.sy = 1; p.sx = 1; p.offy = 0; p.offx = 0;
  launch_gemm(s, p);
}

// C[M][N] = A[M][K] @ B[K][N], all row-major f32 flat buffers
static void gemm_plain(hipStream_t s, const float* A, const float* B, float* C,
                       int M, int N, int K) {
  GemmP p{};
  p.A = A; p.B = B; p.bias = nullptr; p.C = C;
  p.M = M; p.N = N; p.K = K; p.Alayout = 0; p.ij = 0;
  p.KD = p.KH = p.KW = 1; p.pd = p.ph = p.pw = 0;
  p.Din = 1; p.Hin = 1; p.Win = N;
  p.Dn = 1; p.Hn = 1; p.Wn = N;
  p.oH = 1; p.oW = N; p.sy = 1; p.sx = 1; p.offy = 0; p.offx = 0;
  launch_gemm(s, p);
}

// ConvTranspose3d kernel=(1,2,2) stride=(1,2,2); w: (Ci,Co,1,2,2)
static void convT_1x2x2(hipStream_t s, const float* x, int Ci, int D, int H, int W,
                        const float* w, const float* b, int Co, float* y) {
  for (int ij = 0; ij < 4; ++ij) {
    GemmP p{};
    p.A = w; p.B = x; p.bias = b; p.C = y;
    p.M = Co; p.N = D * H * W; p.K = Ci;
    p.Alayout = 1; p.ij = ij;
    p.KD = p.KH = p.KW = 1; p.pd = p.ph = p.pw = 0;
    p.Din = D; p.Hin = H; p.Win = W;
    p.Dn = D; p.Hn = H; p.Wn = W;
    p.oH = 2 * H; p.oW = 2 * W; p.sy = 2; p.sx = 2; p.offy = ij >> 1; p.offx = ij & 1;
    launch_gemm(s, p);
  }
}

static void groupnorm(hipStream_t s, const float* xin, float* y, const float* g,
                      const float* b, int C, long long S, int lrelu, float* stats) {
  gn_reduce<<<4, 256, 0, s>>>(xin, stats, C, S, 4);
  long long total = (long long)C * S;
  gn_apply<<<(int)((total + 255) / 256), 256, 0, s>>>(xin, y, stats, g, b, C, S, 4, lrelu);
}

static void maxpool(hipStream_t s, const float* x, float* y, int C, int D, int H, int W) {
  long long total = (long long)C * D * (H / 2) * (W / 2);
  maxpool_1x2x2<<<(int)((total + 255) / 256), 256, 0, s>>>(x, y, total, D, H, W);
}

static void add2(hipStream_t s, float* dst, const float* a, const float* b, long long n) {
  add2_kernel<<<(int)((n + 255) / 256), 256, 0, s>>>(dst, a, b, n);
}

struct Blk { const float *w1, *b1, *g1, *bb1, *w2, *b2, *g2, *bb2; };
struct Enc { Blk dc1, dc2, dc3, bot; };
struct Dims { int D, H, W; };

static void run_block(hipStream_t s, const float* x, int Ci, Dims d, const Blk& bk,
                      int C1, int C2, float* tmp, float* out, float* stats) {
  long long S = (long long)d.D * d.H * d.W;
  conv3d(s, x, Ci, d.D, d.H, d.W, bk.w1, bk.b1, C1, 1, 3, 3, 0, 1, 1, tmp);
  groupnorm(s, tmp, tmp, bk.g1, bk.bb1, C1, S, 1, stats);
  conv3d(s, tmp, C1, d.D, d.H, d.W, bk.w2, bk.b2, C2, 3, 3, 3, 1, 1, 1, out);
  groupnorm(s, out, out, bk.g2, bk.bb2, C2, S, 1, stats);
}

static void run_encoder(hipStream_t s, const float* x, const Enc& e,
                        float* d1, float* d2, float* d3, float* bot,
                        WS& ws, float* stats) {
  Dims L0{8, 160, 160}, L1{8, 80, 80}, L2{8, 40, 40}, L3{8, 20, 20};
  size_t mk = ws.cur;
  float* t = wsalloc(ws, 32LL * 204800);
  run_block(s, x, 1, L0, e.dc1, 32, 64, t, d1, stats);
  ws.cur = mk;
  float* s1 = wsalloc(ws, 64LL * 51200);
  maxpool(s, d1, s1, 64, 8, 160, 160);
  mk = ws.cur;
  t = wsalloc(ws, 64LL * 51200);
  run_block(s, s1, 64, L1, e.dc2, 64, 128, t, d2, stats);
  ws.cur = mk;
  float* s2 = wsalloc(ws, 128LL * 12800);
  maxpool(s, d2, s2, 128, 8, 80, 80);
  mk = ws.cur;
  t = wsalloc(ws, 128LL * 12800);
  run_block(s, s2, 128, L2, e.dc3, 128, 256, t, d3, stats);
  ws.cur = mk;
  float* s3 = wsalloc(ws, 256LL * 3200);
  maxpool(s, d3, s3, 256, 8, 20, 20);
  mk = ws.cur;
  t = wsalloc(ws, 256LL * 3200);
  run_block(s, s3, 256, L3, e.bot, 256, 512, t, bot, stats);
  ws.cur = mk;
}

} // namespace

// ---------------------------------------------------------------------------
// kernel_launch: full forward pass
// ---------------------------------------------------------------------------
extern "C" void kernel_launch(void* const* d_in, const int* in_sizes, int n_in,
                              void* d_out, int out_size, void* d_ws, size_t ws_size,
                              hipStream_t stream) {
  (void)in_sizes; (void)n_in; (void)out_size; (void)ws_size;
  hipStream_t s = stream;

  int cur = 0;
  auto nxt = [&]() -> const float* { return (const float*)d_in[cur++]; };

  const float* x = nxt();  // (1,1,8,160,160)

  Enc glob, loc;
  auto readBlk = [&](Blk& b) {
    b.w1 = nxt(); b.b1 = nxt(); b.g1 = nxt(); b.bb1 = nxt();
    b.w2 = nxt(); b.b2 = nxt(); b.g2 = nxt(); b.bb2 = nxt();
  };
  auto readEnc = [&](Enc& e) { readBlk(e.dc1); readBlk(e.dc2); readBlk(e.dc3); readBlk(e.bot); };
  readEnc(glob);
  readEnc(loc);
  const float* sk_w = nxt(); const float* sk_b = nxt();
  const float* sq_w = nxt(); const float* sq_b = nxt();
  const float* sv_w = nxt(); const float* sv_b = nxt();
  const float* so_w = nxt(); const float* so_b = nxt();
  const float* gn_g = nxt(); const float* gn_b = nxt();
  const float* sc_w = nxt(); const float* sc_b = nxt();
  const float* ck_w = nxt(); const float* ck_b = nxt();
  const float* cq_w = nxt(); const float* cq_b = nxt();
  const float* cv_w = nxt(); const float* cv_b = nxt();
  const float* co_w = nxt(); const float* co_b = nxt();
  const float* cc_w = nxt(); const float* cc_b = nxt();
  const float* up1_w = nxt(); const float* up1_b = nxt();
  Blk upc1; readBlk(upc1);
  const float* up2_w = nxt(); const float* up2_b = nxt();
  Blk upc2; readBlk(upc2);
  const float* up3_w = nxt(); const float* up3_b = nxt();
  Blk upc3; readBlk(upc3);
  const float* out_w = nxt(); const float* out_b = nxt();

  WS ws{(char*)d_ws, 0, ws_size};

  // Persistent buffers
  float* ld1  = wsalloc(ws, 64LL * 204800);
  float* ld2  = wsalloc(ws, 128LL * 51200);
  float* ld3  = wsalloc(ws, 256LL * 12800);
  float* lbot = wsalloc(ws, 512LL * 3200);
  float* gbot = wsalloc(ws, 512LL * 3200);
  float* bott = wsalloc(ws, 1024LL * 3200);
  float* att1  = bott;
  float* catt1 = bott + 512LL * 3200;
  float* stats = wsalloc(ws, 64);

  // ---- encoders ----
  size_t mk0 = ws.cur;
  float* gd1 = wsalloc(ws, 64LL * 204800);
  float* gd2 = wsalloc(ws, 128LL * 51200);
  float* gd3 = wsalloc(ws, 256LL * 12800);
  run_encoder(s, x, glob, gd1, gd2, gd3, gbot, ws, stats);
  ws.cur = mk0;
  run_encoder(s, x, loc, ld1, ld2, ld3, lbot, ws, stats);

  const int S3 = 3200;  // 8*20*20

  // ---- self-attention at bottleneck ----
  size_t mkA = ws.cur;
  float* kb   = wsalloc(ws, 256LL * S3);
  float* qb   = wsalloc(ws, 256LL * S3);
  float* vb   = wsalloc(ws, 256LL * S3);
  float* Smat = wsalloc(ws, (long long)S3 * S3);
  float* attv = wsalloc(ws, 256LL * S3);
  float* t512a = wsalloc(ws, 512LL * S3);
  float* t512b = wsalloc(ws, 512LL * S3);

  conv3d(s, lbot, 512, 8, 20, 20, sk_w, sk_b, 256, 1, 1, 1, 0, 0, 0, kb);
  conv3d(s, lbot, 512, 8, 20, 20, sq_w, sq_b, 256, 1, 1, 1, 0, 0, 0, qb);
  conv3d(s, lbot, 512, 8, 20, 20, sv_w, sv_b, 256, 1, 1, 1, 0, 0, 0, vb);
  // k(3200,256) @ q(256,3200) -- raw-reshape semantics == flat reinterpretation
  gemm_plain(s, kb, qb, Smat, S3, S3, 256);
  softmax_rows<<<S3, 256, 0, s>>>(Smat, S3);
  gemm_plain(s, Smat, vb, attv, S3, 256, S3);
  conv3d(s, attv, 256, 8, 20, 20, so_w, so_b, 512, 1, 1, 1, 0, 0, 0, t512a);
  add2(s, t512a, t512a, lbot, 512LL * S3);
  groupnorm(s, t512a, t512a, gn_g, gn_b, 512, S3, 0, stats);          // att
  conv3d(s, t512a, 512, 8, 20, 20, sc_w, sc_b, 512, 3, 3, 3, 1, 1, 1, t512b);
  add2(s, t512b, t512b, t512a, 512LL * S3);
  groupnorm(s, t512b, att1, gn_g, gn_b, 512, S3, 0, stats);           // att1

  // ---- cross-attention ----
  conv3d(s, gbot, 512, 8, 20, 20, ck_w, ck_b, 256, 1, 1, 1, 0, 0, 0, kb);
  conv3d(s, att1, 512, 8, 20, 20, cq_w, cq_b, 256, 1, 1, 1, 0, 0, 0, qb);
  conv3d(s, att1, 512, 8, 20, 20, cv_w, cv_b, 256, 1, 1, 1, 0, 0, 0, vb);
  gemm_plain(s, kb, qb, Smat, S3, S3, 256);
  softmax_rows<<<S3, 256, 0, s>>>(Smat, S3);
  gemm_plain(s, Smat, vb, attv, S3, 256, S3);
  conv3d(s, attv, 256, 8, 20, 20, co_w, co_b, 512, 1, 1, 1, 0, 0, 0, t512a);
  add2(s, t512a, t512a, att1, 512LL * S3);
  groupnorm(s, t512a, t512a, gn_g, gn_b, 512, S3, 0, stats);          // catt
  conv3d(s, t512a, 512, 8, 20, 20, cc_w, cc_b, 512, 3, 3, 3, 1, 1, 1, t512b);
  add2(s, t512b, t512b, t512a, 512LL * S3);
  groupnorm(s, t512b, catt1, gn_g, gn_b, 512, S3, 0, stats);          // catt1
  ws.cur = mkA;

  // ---- decoder stage 1: bott(1024,8,20,20) -> (8,40,40) ----
  float* u1c = wsalloc(ws, 256LL * 12800);
  size_t mkD = ws.cur;
  {
    float* cat1 = wsalloc(ws, 768LL * 12800);
    float* tD   = wsalloc(ws, 256LL * 12800);
    hipMemcpyAsync(cat1, ld3, 256LL * 12800 * 4, hipMemcpyDeviceToDevice, s);
    convT_1x2x2(s, bott, 1024, 8, 20, 20, up1_w, up1_b, 512, cat1 + 256LL * 12800);
    run_block(s, cat1, 768, Dims{8, 40, 40}, upc1, 256, 256, tD, u1c, stats);
  }
  ws.cur = mkD;

  // ---- decoder stage 2: -> (8,80,80) ----
  float* u2c = wsalloc(ws, 128LL * 51200);
  mkD = ws.cur;
  {
    float* cat2 = wsalloc(ws, 384LL * 51200);
    float* tD   = wsalloc(ws, 128LL * 51200);
    hipMemcpyAsync(cat2, ld2, 128LL * 51200 * 4, hipMemcpyDeviceToDevice, s);
    convT_1x2x2(s, u1c, 256, 8, 40, 40, up2_w, up2_b, 256, cat2 + 128LL * 51200);
    run_block(s, cat2, 384, Dims{8, 80, 80}, upc2, 128, 128, tD, u2c, stats);
  }
  ws.cur = mkD;

  // ---- decoder stage 3: -> (8,160,160) ----
  float* u3c = wsalloc(ws, 64LL * 204800);
  mkD = ws.cur;
  {
    float* cat3 = wsalloc(ws, 192LL * 204800);
    float* tD   = wsalloc(ws, 64LL * 204800);
    hipMemcpyAsync(cat3, ld1, 64LL * 204800 * 4, hipMemcpyDeviceToDevice, s);
    convT_1x2x2(s, u2c, 128, 8, 80, 80, up3_w, up3_b, 128, cat3 + 64LL * 204800);
    run_block(s, cat3, 192, Dims{8, 160, 160}, upc3, 64, 64, tD, u3c, stats);
  }
  ws.cur = mkD;

  // ---- output head: 1x1x1 conv 64->4, channel softmax ----
  float* logits = wsalloc(ws, 4LL * 204800);
  conv3d(s, u3c, 64, 8, 160, 160, out_w, out_b, 4, 1, 1, 1, 0, 0, 0, logits);
  float* outp = (float*)d_out;
  softmax_c4<<<(204800 + 255) / 256, 256, 0, s>>>(logits, outp, 204800);

  // outputs: (seg, gbot, lbot) concatenated flat
  hipMemcpyAsync(outp + 819200, gbot, 512LL * 3200 * 4, hipMemcpyDeviceToDevice, s);
  hipMemcpyAsync(outp + 819200 + 1638400, lbot, 512LL * 3200 * 4, hipMemcpyDeviceToDevice, s);
}
